// RGA_direct_with_charge_interac_acfct_encode_whole_15401752723473
// MI455X (gfx1250) — compile-verified, matrix operand order checked by reading, not run on hardware
//
#include <hip/hip_runtime.h>
#include <hip/hip_bf16.h>
#include <string.h>

typedef _Float16 v16h __attribute__((ext_vector_type(16)));
typedef _Float16 v4h  __attribute__((ext_vector_type(4)));
typedef float    v8f  __attribute__((ext_vector_type(8)));

#define CDIV(a,b) (((a)+(b)-1)/(b))

// ---------- float <-> order-preserving uint (for atomic max on floats) ----------
__device__ __forceinline__ unsigned f2o(float f) {
  unsigned u = __float_as_uint(f);
  return (u & 0x80000000u) ? ~u : (u | 0x80000000u);
}
__device__ __forceinline__ float o2f(unsigned k) {
  unsigned u = (k & 0x80000000u) ? (k & 0x7fffffffu) : ~k;
  return __uint_as_float(u);
}
static inline unsigned host_f2o(float f) {
  unsigned u; memcpy(&u, &f, 4);
  return (u & 0x80000000u) ? ~u : (u | 0x80000000u);
}

// ---------------------------------- fills ----------------------------------
__global__ void k_fill_f32(float* p, float v, int n) {
  int i = blockIdx.x * blockDim.x + threadIdx.x;
  if (i < n) p[i] = v;
}
__global__ void k_fill_u32(unsigned* p, unsigned v, int n) {
  int i = blockIdx.x * blockDim.x + threadIdx.x;
  if (i < n) p[i] = v;
}

// ------------------------- batch-norm (training stats) ----------------------
__global__ __launch_bounds__(256) void k_bn_stats(const float* __restrict__ x,
    int n, int d, float* __restrict__ mean, float* __restrict__ rstd) {
  int c = blockIdx.x;                      // one block per feature column
  __shared__ float s1[256], s2[256];
  float a = 0.f, b = 0.f;
  for (int i = threadIdx.x; i < n; i += 256) {
    float v = x[(size_t)i * d + c];
    a += v; b += v * v;
  }
  s1[threadIdx.x] = a; s2[threadIdx.x] = b;
  __syncthreads();
  for (int s = 128; s > 0; s >>= 1) {
    if (threadIdx.x < s) { s1[threadIdx.x] += s1[threadIdx.x + s]; s2[threadIdx.x] += s2[threadIdx.x + s]; }
    __syncthreads();
  }
  if (threadIdx.x == 0) {
    float mu  = s1[0] / (float)n;
    float var = s2[0] / (float)n - mu * mu;
    if (var < 0.f) var = 0.f;
    mean[c] = mu;
    rstd[c] = rsqrtf(var + 1e-5f);
  }
}

// y[i, yoff+c (stride yld)] = g*(x-mu)*rstd + b  (+res)  (relu?)
__global__ void k_bn_apply(float* __restrict__ y, const float* __restrict__ x,
    const float* __restrict__ res, const float* __restrict__ g, const float* __restrict__ b,
    const float* __restrict__ mean, const float* __restrict__ rstd,
    int n, int d, int yld, int yoff, int relu) {
  int idx = blockIdx.x * blockDim.x + threadIdx.x;
  if (idx >= n * d) return;
  int i = idx / d, c = idx % d;
  float v = g[c] * (x[idx] - mean[c]) * rstd[c] + b[c];
  if (res) v += res[idx];
  if (relu && v < 0.f) v = 0.f;
  y[(size_t)i * yld + yoff + c] = v;
}

// ------------------------- small-K dense linear -----------------------------
__global__ void k_simple_gemm(float* __restrict__ y, const float* __restrict__ x,
    const float* __restrict__ W, const float* __restrict__ bias,
    int n, int din, int dout, int relu) {
  int idx = blockIdx.x * blockDim.x + threadIdx.x;
  if (idx >= n * dout) return;
  int i = idx / dout, o = idx % dout;
  float acc = bias ? bias[o] : 0.f;
  const float* xr = x + (size_t)i * din;
  const float* wr = W + (size_t)o * din;
  for (int k = 0; k < din; ++k) acc += xr[k] * wr[k];
  if (relu && acc < 0.f) acc = 0.f;
  y[idx] = acc;
}

// ------------------------- WMMA GEMM:  Y = X @ W^T + b ----------------------
// X:[n,din] f32, W:[dout,din] f32, Y:[n,dout] f32.  f16 inputs, f32 accumulate.
// Block = 8 waves (2x4) -> 32x64 output tile; K-step 32 staged in LDS.
// Fast path: issue all three b128 tile loads first (one loadcnt wait total),
// then prefetch next K-chunk, then convert + packed ds_store_b64.
// Edge path: clamp addresses (always valid), select value on K-guard only.
__global__ __launch_bounds__(256) void k_wmma_gemm(float* __restrict__ Y,
    const float* __restrict__ X, const float* __restrict__ W,
    const float* __restrict__ bias, int n, int din, int dout, int relu) {
  __shared__ _Float16 As[32][40];   // X tile  (rows x k)
  __shared__ _Float16 Bs[64][40];   // W rows  (out-cols x k)
  const int tid  = threadIdx.x;
  const int lane = tid & 31;
  const int wave = tid >> 5;
  const int wm   = wave >> 2;       // 0..1
  const int wn   = wave & 3;        // 0..3
  const int rowbase = blockIdx.x * 32;
  const int colbase = blockIdx.y * 64;
  const bool interior = (rowbase + 32 <= n) && (colbase + 64 <= dout);
  v8f acc = {};
  for (int k0 = 0; k0 < din; k0 += 32) {
    if (interior && (k0 + 32 <= din)) {
      // ---- fast path: batch all global loads, single wait, then convert ----
      const int r  = tid >> 3;            // 0..31
      const int c4 = (tid & 7) * 4;       // 0,4,...,28
      const float* sA  = X + (size_t)(rowbase + r)      * din + k0 + c4;
      const float* sB0 = W + (size_t)(colbase + r)      * din + k0 + c4;
      const float* sB1 = W + (size_t)(colbase + r + 32) * din + k0 + c4;
      float a0 = sA[0],  a1 = sA[1],  a2 = sA[2],  a3 = sA[3];
      float b0 = sB0[0], b1 = sB0[1], b2 = sB0[2], b3 = sB0[3];
      float c0 = sB1[0], c1 = sB1[1], c2 = sB1[2], c3 = sB1[3];
      if (k0 + 32 < din) {
        __builtin_prefetch(sA  + 32, 0, 1);
        __builtin_prefetch(sB0 + 32, 0, 1);
        __builtin_prefetch(sB1 + 32, 0, 1);
      }
      v4h ha = { (_Float16)a0, (_Float16)a1, (_Float16)a2, (_Float16)a3 };
      v4h hb = { (_Float16)b0, (_Float16)b1, (_Float16)b2, (_Float16)b3 };
      v4h hc = { (_Float16)c0, (_Float16)c1, (_Float16)c2, (_Float16)c3 };
      *(v4h*)&As[r][c4]      = ha;
      *(v4h*)&Bs[r][c4]      = hb;
      *(v4h*)&Bs[r + 32][c4] = hc;
    } else {
      // ---- edge path: clamp addresses, select value on K-guard ----
#pragma unroll
      for (int t = tid; t < 32 * 32; t += 256) {
        int r = t >> 5, c = t & 31;
        int gr = rowbase + r; gr = (gr < n) ? gr : (n - 1);
        int gc = k0 + c;
        int gcc = (gc < din) ? gc : (din - 1);
        float v = X[(size_t)gr * din + gcc];
        As[r][c] = (_Float16)((gc < din) ? v : 0.f);
      }
#pragma unroll
      for (int t = tid; t < 64 * 32; t += 256) {
        int r = t >> 5, c = t & 31;
        int gr = colbase + r; gr = (gr < dout) ? gr : (dout - 1);
        int gc = k0 + c;
        int gcc = (gc < din) ? gc : (din - 1);
        float v = W[(size_t)gr * din + gcc];
        Bs[r][c] = (_Float16)((gc < din) ? v : 0.f);
      }
    }
    __syncthreads();
    // Pack fragments per ISA 7.12.2 16-bit 16x32 layout:
    // lanes 0-15: K=0..7 & 16..23 ; lanes 16-31: +8
    const int mrow = wm * 16 + (lane & 15);
    const int nrow = wn * 16 + (lane & 15);
    const int koff = (lane >= 16) ? 8 : 0;
    v16h a, b;
#pragma unroll
    for (int j = 0; j < 16; ++j) {
      int v  = j >> 1, hl = j & 1;
      int kk = ((v < 4) ? (2 * v) : (16 + 2 * (v - 4))) + hl + koff;
      a[j] = As[mrow][kk];
      b[j] = Bs[nrow][kk];
    }
    acc = __builtin_amdgcn_wmma_f32_16x16x32_f16(false, a, false, b, (short)0, acc, false, false);
    __syncthreads();
  }
  // C/D layout: VGPR r -> M = r (+8 for lanes 16-31), N = lane&15
  const int col = colbase + wn * 16 + (lane & 15);
  if (col < dout) {
    float bv = bias ? bias[col] : 0.f;
#pragma unroll
    for (int r = 0; r < 8; ++r) {
      int row = rowbase + wm * 16 + r + ((lane >= 16) ? 8 : 0);
      if (row < n) {
        float v = acc[r] + bv;
        if (relu && v < 0.f) v = 0.f;
        Y[(size_t)row * dout + col] = v;
      }
    }
  }
}

// ---------------------- GATv2 graph kernels --------------------------------
// self-loop attr: loop_attr[v] = sum_{e: dst=v} ea[e] / max(deg,1)
__global__ void k_deg_loop1(float* __restrict__ deg, float* __restrict__ loop,
    const float* __restrict__ ea, const int* __restrict__ dst, int E, int edim) {
  int e = blockIdx.x * blockDim.x + threadIdx.x;
  if (e >= E) return;
  int t = dst[e];
  atomicAdd(&deg[t], 1.f);
  for (int k = 0; k < edim; ++k)
    atomicAdd(&loop[(size_t)t * edim + k], ea[(size_t)e * edim + k]);
}
__global__ void k_deg_loop2(float* __restrict__ loop, const float* __restrict__ deg,
    int n, int edim) {
  int idx = blockIdx.x * blockDim.x + threadIdx.x;
  if (idx >= n * edim) return;
  loop[idx] /= fmaxf(deg[idx / edim], 1.f);
}

// per (ext-edge, head): logits + atomic segment-max over dst
__global__ void k_gat_logits(float* __restrict__ logit, unsigned* __restrict__ lmaxu,
    const float* __restrict__ xl, const float* __restrict__ xr,
    const float* __restrict__ att, const float* __restrict__ We,
    const float* __restrict__ ea, const float* __restrict__ loopattr, int edim,
    const int* __restrict__ src, const int* __restrict__ dst, int E, int n, int H, int C) {
  int idx = blockIdx.x * blockDim.x + threadIdx.x;
  int EN = E + n;
  if (idx >= EN * H) return;
  int e = idx / H, h = idx % H;
  int s, t; const float* eap = nullptr;
  if (e < E) { s = src[e]; t = dst[e]; if (We) eap = ea + (size_t)e * edim; }
  else       { s = t = e - E;          if (We) eap = loopattr + (size_t)s * edim; }
  int HC = H * C;
  const float* xls = xl + (size_t)s * HC + h * C;
  const float* xrt = xr + (size_t)t * HC + h * C;
  const float* ath = att + h * C;
  float acc = 0.f;
  for (int c = 0; c < C; ++c) {
    float m = xls[c] + xrt[c];
    if (We) {
      const float* wrow = We + (size_t)(h * C + c) * edim;
      float em = 0.f;
      for (int k = 0; k < edim; ++k) em += eap[k] * wrow[k];
      m += em;
    }
    m = (m > 0.f) ? m : 0.2f * m;           // leaky_relu(0.2)
    acc += m * ath[c];
  }
  logit[idx] = acc;
  atomicMax(&lmaxu[t * H + h], f2o(acc));
}

// a = exp(logit - lmax[dst]); denom[dst] += a   (in-place on logit buffer)
__global__ void k_gat_expsum(float* __restrict__ a, const unsigned* __restrict__ lmaxu,
    float* __restrict__ den, const int* __restrict__ dst, int E, int n, int H) {
  int idx = blockIdx.x * blockDim.x + threadIdx.x;
  int EN = E + n;
  if (idx >= EN * H) return;
  int e = idx / H, h = idx % H;
  int t = (e < E) ? dst[e] : (e - E);
  float v = __expf(a[idx] - o2f(lmaxu[t * H + h]));
  a[idx] = v;
  atomicAdd(&den[t * H + h], v);
}

// out[dst,h,c] += (a/den[dst,h]) * xl[src,h,c]
__global__ void k_gat_accum(float* __restrict__ out, const float* __restrict__ a,
    const float* __restrict__ den, const float* __restrict__ xl,
    const int* __restrict__ src, const int* __restrict__ dst, int E, int n, int H, int C) {
  int HC = H * C;
  long idx = (long)blockIdx.x * blockDim.x + threadIdx.x;
  long tot = (long)(E + n) * HC;
  if (idx >= tot) return;
  int e  = (int)(idx / HC);
  int hc = (int)(idx % HC);
  int h  = hc / C;
  int s, t;
  if (e < E) { s = src[e]; t = dst[e]; } else { s = t = e - E; }
  float alpha = a[(size_t)e * H + h] / den[t * H + h];
  atomicAdd(&out[(size_t)t * HC + hc], alpha * xl[(size_t)s * HC + hc]);
}

// y[i,c] = mean_h out[i,h,c] + bias[c]
__global__ void k_gat_headmean(float* __restrict__ y, const float* __restrict__ out,
    const float* __restrict__ bias, int n, int H, int C) {
  int idx = blockIdx.x * blockDim.x + threadIdx.x;
  if (idx >= n * C) return;
  int i = idx / C, c = idx % C;
  int HC = H * C;
  float s = 0.f;
  for (int h = 0; h < H; ++h) s += out[(size_t)i * HC + h * C + c];
  y[idx] = s / (float)H + bias[c];
}

// ---------------------- pair-graph construction / gather --------------------
__global__ void k_copy_pairs(int* __restrict__ csrc, int* __restrict__ cdst,
    const int* __restrict__ pairs, int M, int off, int flip) {
  int m = blockIdx.x * blockDim.x + threadIdx.x;
  if (m >= M) return;
  int a = pairs[m], b = pairs[M + m];
  csrc[off + m] = flip ? b : a;
  cdst[off + m] = flip ? a : b;
}

__global__ void k_gather_pair(float* __restrict__ feat, const float* __restrict__ latent,
    const int* __restrict__ idx, int M, float fill) {
  long t = (long)blockIdx.x * blockDim.x + threadIdx.x;
  if (t >= (long)M * 257) return;
  int m = (int)(t / 257), c = (int)(t % 257);
  float v;
  if (c < 128)       v = latent[(size_t)idx[m] * 128 + c];
  else if (c < 256)  v = latent[(size_t)idx[M + m] * 128 + (c - 128)];
  else               v = fill;
  feat[t] = v;
}

// ============================ host orchestration ============================
struct BN  { const float *b, *g; };
struct Lin { const float *W, *b; };
struct Gat { const float *att, *bias, *We; Lin ll, lr; };
struct ResPair { BN bn1, bn5; Lin fc1; BN rbn[3]; Lin rfc[3]; Lin fc5, fc6; };

extern "C" void kernel_launch(void* const* d_in, const int* in_sizes, int n_in,
                              void* d_out, int out_size, void* d_ws, size_t ws_size,
                              hipStream_t stream) {
  // -------- inputs --------
  const float* X    = (const float*)d_in[0];
  const int*   EIDX = (const int*)  d_in[1];
  const float* EA   = (const float*)d_in[2];
  const float* ACF  = (const float*)d_in[3];
  const float* CHG  = (const float*)d_in[4];
  const int* BIDX = (const int*)d_in[5];
  const int* AIDX = (const int*)d_in[6];
  const int* DIDX = (const int*)d_in[7];
  const int* HIDX = (const int*)d_in[8];

  const int N  = in_sizes[0] / 39;
  const int E  = in_sizes[1] / 2;
  const int NB = in_sizes[5] / 2, NA = in_sizes[6] / 2;
  const int ND = in_sizes[7] / 2, NH = in_sizes[8] / 2;
  const int CE = 2 * (NB + NA + ND + NH);
  const int Mmax = (NA > NB ? (ND > NA ? (NH > ND ? NH : ND) : (NH > NA ? NH : NA))
                            : (ND > NB ? (NH > ND ? NH : ND) : (NH > NB ? NH : NB)));

  // -------- params: jax pytree flatten order (sorted dict keys) --------
  int pi = 9;
  auto nextF  = [&]() { return (const float*)d_in[pi++]; };
  auto readBN = [&](BN& o)  { o.b = nextF(); o.g = nextF(); };
  auto readLin= [&](Lin& o, bool bias = true) { o.W = nextF(); o.b = bias ? nextF() : nullptr; };
  auto readGat= [&](Gat& o, bool edge) {
    o.att = nextF(); o.bias = nextF();
    o.We = edge ? nextF() : nullptr;      // lin_e (no bias)
    readLin(o.ll); readLin(o.lr);
  };
  auto readRP = [&](ResPair& o) {
    readBN(o.bn1); readBN(o.bn5); readLin(o.fc1);
    for (int i = 0; i < 3; ++i) { readBN(o.rbn[i]); readLin(o.rfc[i]); }   // fc2..fc4: {bn,fc}
    readLin(o.fc5); readLin(o.fc6);
  };

  BN acfct_bn1; readBN(acfct_bn1);
  BN acfct_bn2; readBN(acfct_bn2);
  BN acb1; readBN(acb1);           // acfct_conv_bn1
  BN acb2; readBN(acb2);
  BN acb3; readBN(acb3);
  Lin acfc1; readLin(acfc1);       // acfct_conv_fc1 (7->128)
  Lin acfc3; readLin(acfc3);       // acfct_conv_fc3 (128->32)
  Gat acgat; readGat(acgat, false);
  Lin fn1; readLin(fn1);           // acfct_fn1 (7->32)
  BN fn2bn; readBN(fn2bn); Lin fn2fc; readLin(fn2fc);   // acfct_fn2
  ResPair angleP; readRP(angleP);
  ResPair bondP;  readRP(bondP);
  BN cbn1; readBN(cbn1); BN cbn2; readBN(cbn2); BN cbn3; readBN(cbn3);
  Lin cfc1; readLin(cfc1); Lin cfc3; readLin(cfc3);
  Gat cgat; readGat(cgat, false);
  ResPair dihP; readRP(dihP);
  BN enbn1; readBN(enbn1); BN enbn5; readBN(enbn5);
  Gat g1; readGat(g1, true);
  BN g2bn; readBN(g2bn); Gat g2; readGat(g2, true);
  BN g3bn; readBN(g3bn); Gat g3; readGat(g3, true);
  BN g4bn; readBN(g4bn); Gat g4; readGat(g4, true);
  Gat g5; readGat(g5, true);
  ResPair hopP; readRP(hopP);
  BN tin; readBN(tin);

  // -------- workspace carve --------
  char* wp = (char*)d_ws;
  auto alloc = [&](size_t bytes) { char* p = wp; wp += (bytes + 255) & ~(size_t)255; return p; };
  float* F_h    = (float*)alloc((size_t)N * 256 * 4);
  float* F_h2   = (float*)alloc((size_t)N * 256 * 4);
  float* F_xl   = (float*)alloc((size_t)N * 1024 * 4);
  float* F_xr   = (float*)alloc((size_t)N * 1024 * 4);
  float* F_hc   = (float*)alloc((size_t)N * 1024 * 4);
  int ENmax = ((E > CE ? E : CE) + N);
  float*    F_a    = (float*)   alloc((size_t)ENmax * 4 * 4);
  unsigned* U_lmax = (unsigned*)alloc((size_t)N * 4 * 4);
  float*    F_den  = (float*)   alloc((size_t)N * 4 * 4);
  float*    F_loop = (float*)   alloc((size_t)N * 10 * 4);
  float*    F_deg  = (float*)   alloc((size_t)N * 4);
  float*    F_mean = (float*)   alloc(1024 * 4);
  float*    F_rstd = (float*)   alloc(1024 * 4);
  float* F_t39  = (float*)alloc((size_t)N * 39 * 4);
  float* F_t128 = (float*)alloc((size_t)N * 128 * 4);
  float* F_t128b= (float*)alloc((size_t)N * 128 * 4);
  float* F_n32a = (float*)alloc((size_t)N * 32 * 4);
  float* F_n32b = (float*)alloc((size_t)N * 32 * 4);
  int* I_cis = (int*)alloc((size_t)CE * 4);
  int* I_cid = (int*)alloc((size_t)CE * 4);
  float* F_pf = (float*)alloc((size_t)Mmax * 257 * 4);
  float* F_px = (float*)alloc((size_t)Mmax * 256 * 4);
  float* F_py = (float*)alloc((size_t)Mmax * 256 * 4);

  float* latent = (float*)d_out;                     // [N,128]
  float* headOut = latent + (size_t)N * 128;

  const int* gsrc = EIDX;
  const int* gdst = EIDX + E;
  const unsigned NEGINF = host_f2o(-3.0e38f);

  // -------- helpers --------
  auto gemm = [&](float* Y, const float* Xp, const Lin& L, int n, int din, int dout, int relu) {
    if (din >= 64)
      k_wmma_gemm<<<dim3(CDIV(n, 32), CDIV(dout, 64)), 256, 0, stream>>>(Y, Xp, L.W, L.b, n, din, dout, relu);
    else
      k_simple_gemm<<<CDIV(n * dout, 256), 256, 0, stream>>>(Y, Xp, L.W, L.b, n, din, dout, relu);
  };
  auto bn_stats = [&](const float* x, int n, int d) {
    k_bn_stats<<<d, 256, 0, stream>>>(x, n, d, F_mean, F_rstd);
  };
  auto bn_apply = [&](float* y, const float* x, const float* res, const BN& p,
                      int n, int d, int yld, int yoff, int relu) {
    k_bn_apply<<<CDIV(n * d, 256), 256, 0, stream>>>(y, x, res, p.g, p.b, F_mean, F_rstd,
                                                     n, d, yld, yoff, relu);
  };
  auto run_gat = [&](const float* hin, int din, int H, int C, const Gat& g,
                     const int* src, const int* dst, int Eg, const float* ea, int edim,
                     float* out) {
    int HC = H * C;
    Lin ll{g.ll.W, g.ll.b}, lr{g.lr.W, g.lr.b};
    gemm(F_xl, hin, ll, N, din, HC, 0);
    gemm(F_xr, hin, lr, N, din, HC, 0);
    const float* loopattr = nullptr;
    if (ea) {
      k_fill_f32<<<CDIV(N, 256), 256, 0, stream>>>(F_deg, 0.f, N);
      k_fill_f32<<<CDIV(N * edim, 256), 256, 0, stream>>>(F_loop, 0.f, N * edim);
      k_deg_loop1<<<CDIV(Eg, 256), 256, 0, stream>>>(F_deg, F_loop, ea, dst, Eg, edim);
      k_deg_loop2<<<CDIV(N * edim, 256), 256, 0, stream>>>(F_loop, F_deg, N, edim);
      loopattr = F_loop;
    }
    k_fill_u32<<<CDIV(N * H, 256), 256, 0, stream>>>(U_lmax, NEGINF, N * H);
    k_fill_f32<<<CDIV(N * H, 256), 256, 0, stream>>>(F_den, 0.f, N * H);
    k_fill_f32<<<CDIV(N * HC, 256), 256, 0, stream>>>(F_hc, 0.f, N * HC);
    int ENg = Eg + N;
    k_gat_logits<<<CDIV(ENg * H, 256), 256, 0, stream>>>(F_a, U_lmax, F_xl, F_xr, g.att, g.We,
                                                         ea, loopattr, edim, src, dst, Eg, N, H, C);
    k_gat_expsum<<<CDIV(ENg * H, 256), 256, 0, stream>>>(F_a, U_lmax, F_den, dst, Eg, N, H);
    long tot = (long)ENg * HC;
    k_gat_accum<<<(int)CDIV(tot, 256), 256, 0, stream>>>(F_hc, F_a, F_den, F_xl, src, dst, Eg, N, H, C);
    k_gat_headmean<<<CDIV(N * C, 256), 256, 0, stream>>>(out, F_hc, g.bias, N, H, C);
  };
  auto run_respair = [&](const int* idx, int M, float fill, const ResPair& rp, float* outp) {
    k_gather_pair<<<(int)CDIV((long)M * 257, 256), 256, 0, stream>>>(F_pf, latent, idx, M, fill);
    gemm(F_px, F_pf, rp.fc1, M, 257, 256, 0);
    bn_stats(F_px, M, 256);
    bn_apply(F_px, F_px, nullptr, rp.bn1, M, 256, 256, 0, 1);
    for (int i = 0; i < 3; ++i) {
      gemm(F_py, F_px, rp.rfc[i], M, 256, 256, 0);
      bn_stats(F_py, M, 256);
      bn_apply(F_px, F_py, F_px, rp.rbn[i], M, 256, 256, 0, 1);  // relu(bn(fc(x)) + x)
    }
    gemm(F_py, F_px, rp.fc5, M, 256, 32, 0);
    bn_stats(F_py, M, 32);
    bn_apply(F_py, F_py, nullptr, rp.bn5, M, 32, 32, 0, 1);
    k_simple_gemm<<<CDIV(M, 256), 256, 0, stream>>>(outp, F_py, rp.fc6.W, rp.fc6.b, M, 32, 1, 1);
  };

  // ================== topo encoder ==================
  bn_stats(X, N, 39);
  bn_apply(F_t39, X, nullptr, tin, N, 39, 39, 0, 0);

  run_gat(F_t39, 39, 4, 256, g1, gsrc, gdst, E, EA, 10, F_h2);
  bn_stats(F_h2, N, 256);
  bn_apply(F_h, F_h2, nullptr, enbn1, N, 256, 256, 0, 1);

  const Gat* gl[3] = { &g2, &g3, &g4 };
  const BN*  gb[3] = { &g2bn, &g3bn, &g4bn };
  for (int L = 0; L < 3; ++L) {
    run_gat(F_h, 256, 4, 256, *gl[L], gsrc, gdst, E, EA, 10, F_h2);
    bn_stats(F_h2, N, 256);
    bn_apply(F_h, F_h2, F_h, *gb[L], N, 256, 256, 0, 1);         // relu(bn(gat)+h)
  }
  run_gat(F_h, 256, 4, 32, g5, gsrc, gdst, E, EA, 10, F_n32a);
  bn_stats(F_n32a, N, 32);
  bn_apply(latent, F_n32a, nullptr, enbn5, N, 32, 128, 0, 1);    // latent[:, 0:32]

  // ================== charge-interaction graph ==================
  {
    int off = 0;
    k_copy_pairs<<<CDIV(NB,256),256,0,stream>>>(I_cis, I_cid, BIDX, NB, off, 0); off += NB;
    k_copy_pairs<<<CDIV(NB,256),256,0,stream>>>(I_cis, I_cid, BIDX, NB, off, 1); off += NB;
    k_copy_pairs<<<CDIV(NA,256),256,0,stream>>>(I_cis, I_cid, AIDX, NA, off, 0); off += NA;
    k_copy_pairs<<<CDIV(NA,256),256,0,stream>>>(I_cis, I_cid, AIDX, NA, off, 1); off += NA;
    k_copy_pairs<<<CDIV(ND,256),256,0,stream>>>(I_cis, I_cid, DIDX, ND, off, 0); off += ND;
    k_copy_pairs<<<CDIV(ND,256),256,0,stream>>>(I_cis, I_cid, DIDX, ND, off, 1); off += ND;
    k_copy_pairs<<<CDIV(NH,256),256,0,stream>>>(I_cis, I_cid, HIDX, NH, off, 0); off += NH;
    k_copy_pairs<<<CDIV(NH,256),256,0,stream>>>(I_cis, I_cid, HIDX, NH, off, 1);
  }

  // ================== charge encoder ==================
  k_simple_gemm<<<CDIV(N*128,256),256,0,stream>>>(F_t128, CHG, cfc1.W, cfc1.b, N, 1, 128, 0);
  bn_stats(F_t128, N, 128);
  bn_apply(F_t128, F_t128, nullptr, cbn1, N, 128, 128, 0, 1);
  run_gat(F_t128, 128, 4, 128, cgat, I_cis, I_cid, CE, nullptr, 0, F_t128b);
  bn_stats(F_t128b, N, 128);
  bn_apply(F_t128b, F_t128b, nullptr, cbn2, N, 128, 128, 0, 1);
  gemm(F_n32a, F_t128b, cfc3, N, 128, 32, 0);
  bn_stats(F_n32a, N, 32);
  bn_apply(latent, F_n32a, nullptr, cbn3, N, 32, 128, 32, 1);    // latent[:, 32:64]

  // ================== acfct conv encoder ==================
  k_simple_gemm<<<CDIV(N*128,256),256,0,stream>>>(F_t128, ACF, acfc1.W, acfc1.b, N, 7, 128, 0);
  bn_stats(F_t128, N, 128);
  bn_apply(F_t128, F_t128, nullptr, acb1, N, 128, 128, 0, 1);
  run_gat(F_t128, 128, 4, 128, acgat, I_cis, I_cid, CE, nullptr, 0, F_t128b);
  bn_stats(F_t128b, N, 128);
  bn_apply(F_t128b, F_t128b, nullptr, acb2, N, 128, 128, 0, 1);
  gemm(F_n32a, F_t128b, acfc3, N, 128, 32, 0);
  bn_stats(F_n32a, N, 32);
  bn_apply(latent, F_n32a, nullptr, acb3, N, 32, 128, 64, 1);    // latent[:, 64:96]

  // ================== acfct encoder ==================
  k_simple_gemm<<<CDIV(N*32,256),256,0,stream>>>(F_n32a, ACF, fn1.W, fn1.b, N, 7, 32, 0);
  bn_stats(F_n32a, N, 32);
  bn_apply(F_n32a, F_n32a, nullptr, acfct_bn1, N, 32, 32, 0, 1);
  k_simple_gemm<<<CDIV(N*32,256),256,0,stream>>>(F_n32b, F_n32a, fn2fc.W, fn2fc.b, N, 32, 32, 0);
  bn_stats(F_n32b, N, 32);
  bn_apply(F_n32b, F_n32b, F_n32a, fn2bn, N, 32, 32, 0, 1);      // res_linear
  bn_stats(F_n32b, N, 32);
  bn_apply(latent, F_n32b, nullptr, acfct_bn2, N, 32, 128, 96, 1); // latent[:, 96:128]

  // ================== pair prediction heads ==================
  run_respair(BIDX, NB, 1.0f, bondP,  headOut);
  run_respair(AIDX, NA, 2.0f, angleP, headOut + NB);
  run_respair(DIDX, ND, 3.0f, dihP,   headOut + NB + NA);
  run_respair(HIDX, NH, 4.0f, hopP,   headOut + NB + NA + ND);
}
